// Aliformer_44375602102816
// MI455X (gfx1250) — matrix operand order repaired and checked by reading, not compiled, not run on hardware
//
#include <hip/hip_runtime.h>
#include <hip/hip_bf16.h>

// ---------------------------------------------------------------------------
// Aliformer forward for MI455X (gfx1250, wave32, WMMA).
// All GEMMs + attention use v_wmma_f32_16x16x32_bf16 with f32 accumulation.
// Tile staging uses CDNA5 async memory->LDS DMA (ASYNCcnt) with LDS double
// buffering so global fetch of tile t+1 overlaps WMMA on tile t.
// ---------------------------------------------------------------------------

#define B_  4
#define L_  1024
#define DM_ 512
#define DF_ 1024
#define H_  8
#define DK_ 128
#define M_  (B_ * L_)          // 4096 rows of the flat activation matrix
#define SLAB_ (L_ * DK_)       // 131072: contiguous (b,h) slab in flat layout

typedef __attribute__((ext_vector_type(16))) __bf16 v16bf;
typedef __attribute__((ext_vector_type(8)))  float  v8f;

union FragB16 { uint4 u[2]; v16bf v; };   // 16 bf16 = 32B = 8 VGPRs

__device__ __forceinline__ unsigned short f2bf(float f) {
  unsigned u = __float_as_uint(f);
  u += 0x7FFFu + ((u >> 16) & 1u);        // round-to-nearest-even
  return (unsigned short)(u >> 16);
}

#define WMMA_BF16(a, b, c) \
  __builtin_amdgcn_wmma_f32_16x16x32_bf16(false, (a), false, (b), (short)0, (c), false, false)

// Low 32 bits of a generic (flat) LDS address == LDS byte offset (aperture
// mapping keeps the low 32 bits); this is what GLOBAL_LOAD_ASYNC_TO_LDS's
// VDST operand expects.
__device__ __forceinline__ unsigned lds_off(const void* p) {
  return (unsigned)(uintptr_t)p;
}

// GLOBAL_LOAD_ASYNC_TO_LDS_B128: 16B memory -> LDS per lane, no VGPR bounce,
// tracked with ASYNCcnt (CDNA5 ISA 10., 15.18.3 opcode 98).
__device__ __forceinline__ void g2l_async_b128(unsigned ldsDst, const void* gSrc) {
  asm volatile("global_load_async_to_lds_b128 %0, %1, off"
               :: "v"(ldsDst), "v"(gSrc) : "memory");
}

template<int N>
__device__ __forceinline__ void wait_async() {
  asm volatile("s_wait_asynccnt %0" :: "i"(N) : "memory");
}

// ---------------------------------------------------------------------------
// f32 -> bf16 elementwise conversion (vectorized x4; all sizes are %4 == 0)
// ---------------------------------------------------------------------------
__global__ __launch_bounds__(256) void cvt_f32_bf16(const float* __restrict__ src,
                                                    unsigned short* __restrict__ dst,
                                                    int n4) {
  int i = blockIdx.x * 256 + threadIdx.x;
  if (i < n4) {
    float4 f = ((const float4*)src)[i];
    ushort4 o;
    o.x = f2bf(f.x); o.y = f2bf(f.y); o.z = f2bf(f.z); o.w = f2bf(f.w);
    ((ushort4*)dst)[i] = o;
  }
}

// ---------------------------------------------------------------------------
// C[M,N] = A[M,K] @ W[N,K]^T + bias.  A,W bf16 row-major; C -> bf16 and/or f32.
// Block: 256 thr (8 waves, 2x4 grid), tile 64x128, each wave 32x32 (2x2 WMMA),
// K-step 32.  Double-buffered LDS fed by async DMA: issue tile t+1, wait for
// tile t (in-order async completion), compute tile t.
// Fragment layouts per CDNA5 ISA 7.12.2 (16-bit A 16x32 / B 32x16).
// ---------------------------------------------------------------------------
__global__ __launch_bounds__(256) void gemm_bf16(const unsigned short* __restrict__ A,
                                                 const unsigned short* __restrict__ W,
                                                 const float* __restrict__ bias,
                                                 unsigned short* __restrict__ Cb,
                                                 float* __restrict__ Cf,
                                                 int M, int N, int K) {
  __shared__ __align__(16) unsigned short lA[2][64 * 32];    // 2 x 4 KB
  __shared__ __align__(16) unsigned short lB[2][128 * 32];   // 2 x 8 KB

  const int tid  = threadIdx.x;
  const int lane = tid & 31, half = lane >> 4, l16 = lane & 15;
  const int wave = tid >> 5, wm = wave & 1, wn = wave >> 1;   // 2 x 4 waves
  const int mBase = blockIdx.y * 64, nBase = blockIdx.x * 128;

  v8f acc[2][2] = {};

  // Per-thread 16B staging chunks
  const int ar  = tid >> 2,         ac  = (tid & 3) << 3;          // A: 64x32
  const int br0 = tid >> 2,         bc0 = (tid & 3) << 3;          // B: rows 0..63
  const int br1 = (tid + 256) >> 2, bc1 = ((tid + 256) & 3) << 3;  // B: rows 64..127

  const unsigned short* aSrc  = &A[(size_t)(mBase + ar)  * K + ac];
  const unsigned short* bSrc0 = &W[(size_t)(nBase + br0) * K + bc0];
  const unsigned short* bSrc1 = &W[(size_t)(nBase + br1) * K + bc1];

  unsigned aDst[2]  = { lds_off(&lA[0][ar  * 32 + ac]),  lds_off(&lA[1][ar  * 32 + ac])  };
  unsigned bDst0[2] = { lds_off(&lB[0][br0 * 32 + bc0]), lds_off(&lB[1][br0 * 32 + bc0]) };
  unsigned bDst1[2] = { lds_off(&lB[0][br1 * 32 + bc1]), lds_off(&lB[1][br1 * 32 + bc1]) };

  const int nT = K >> 5;
  // Prologue: DMA tile 0 into buffer 0
  g2l_async_b128(aDst[0],  aSrc);
  g2l_async_b128(bDst0[0], bSrc0);
  g2l_async_b128(bDst1[0], bSrc1);

  for (int t = 0; t < nT; ++t) {
    const int cur = t & 1, nxt = cur ^ 1;
    if (t + 1 < nT) {                       // DMA tile t+1 while computing tile t
      const int ko = (t + 1) << 5;
      g2l_async_b128(aDst[nxt],  aSrc + ko);
      g2l_async_b128(bDst0[nxt], bSrc0 + ko);
      g2l_async_b128(bDst1[nxt], bSrc1 + ko);
      wait_async<3>();                      // tile t's 3 asyncs complete (in-order)
      if (t + 2 < nT)                       // warm L2 one more tile ahead
        __builtin_prefetch(aSrc + ((t + 2) << 5), 0, 2);
    } else {
      wait_async<0>();
    }
    __syncthreads();

    FragB16 a[2], b[2];
    for (int s = 0; s < 2; ++s) {
      // A 16x32: lanes 0-15 K=0..7 (v0-3) & 16..23 (v4-7); lanes 16-31 +8
      int r = wm * 32 + s * 16 + l16;
      a[s].u[0] = *(const uint4*)&lA[cur][r * 32 + half * 8];
      a[s].u[1] = *(const uint4*)&lA[cur][r * 32 + 16 + half * 8];
      // B 32x16: col = lane%16, lanes 0-15 K=0..7 & 16..23; lanes 16-31 +8
      int n = wn * 32 + s * 16 + l16;
      b[s].u[0] = *(const uint4*)&lB[cur][n * 32 + half * 8];
      b[s].u[1] = *(const uint4*)&lB[cur][n * 32 + 16 + half * 8];
    }
    for (int tm = 0; tm < 2; ++tm)
      for (int tn = 0; tn < 2; ++tn)
        acc[tm][tn] = WMMA_BF16(a[tm].v, b[tn].v, acc[tm][tn]);
    __syncthreads();                        // all waves done reading buf[cur]
  }

  for (int tm = 0; tm < 2; ++tm)
    for (int tn = 0; tn < 2; ++tn) {
      int col = nBase + wn * 32 + tn * 16 + l16;
      float bs = bias ? bias[col] : 0.0f;
      for (int vv = 0; vv < 8; ++vv) {
        int row = mBase + wm * 32 + tm * 16 + vv + half * 8;   // C-layout: M=v(+8)
        float val = acc[tm][tn][vv] + bs;
        if (Cb) Cb[(size_t)row * N + col] = f2bf(val);
        if (Cf) Cf[(size_t)row * N + col] = val;
      }
    }
}

// ---------------------------------------------------------------------------
// Fused dual-score flash attention per contiguous (b,h) slab (1024x128):
//   O = softmax((Q K^T + QF KF^T) / sqrt(2*DK)) V
// Block: 256 thr (8 waves), 128 query rows, key blocks of 64, online softmax.
// K / KF tiles staged by async DMA; V is transposed through VGPRs into LDS
// (overlaps with the in-flight asyncs).
// ---------------------------------------------------------------------------
__global__ __launch_bounds__(256) void attn_fused(const unsigned short* __restrict__ Qg,
                                                  const unsigned short* __restrict__ Kg,
                                                  const unsigned short* __restrict__ QFg,
                                                  const unsigned short* __restrict__ KFg,
                                                  const unsigned short* __restrict__ Vg,
                                                  unsigned short* __restrict__ Og) {
  __shared__ __align__(16) unsigned short lK [64 * 128];   // 16 KB
  __shared__ __align__(16) unsigned short lKF[64 * 128];   // 16 KB
  __shared__ __align__(16) unsigned short lVt[128 * 64];   // 16 KB (V transposed)
  __shared__ __align__(16) unsigned short lP [8 * 16 * 64];// 16 KB (per-wave P)

  const int tid  = threadIdx.x;
  const int lane = tid & 31, half = lane >> 4, l16 = lane & 15;
  const int wave = tid >> 5;
  const size_t slab = (size_t)blockIdx.y * SLAB_;
  const int mBase = blockIdx.x * 128;

  const unsigned short* q  = Qg  + slab;
  const unsigned short* k  = Kg  + slab;
  const unsigned short* qf = QFg + slab;
  const unsigned short* kf = KFg + slab;
  const unsigned short* v  = Vg  + slab;
  unsigned short*       o  = Og  + slab;

  // Persistent Q / QF A-fragments for this wave's 16 query rows (d = 4 k-steps)
  const int mG = mBase + wave * 16 + l16;
  FragB16 qA[4], qfA[4];
  for (int kd = 0; kd < 4; ++kd) {
    int d1 = kd * 32 + half * 8, d2 = kd * 32 + 16 + half * 8;
    qA [kd].u[0] = *(const uint4*)&q [mG * 128 + d1];
    qA [kd].u[1] = *(const uint4*)&q [mG * 128 + d2];
    qfA[kd].u[0] = *(const uint4*)&qf[mG * 128 + d1];
    qfA[kd].u[1] = *(const uint4*)&qf[mG * 128 + d2];
  }

  v8f oacc[8] = {};
  float m_i[8], l_i[8];
  for (int i = 0; i < 8; ++i) { m_i[i] = -3.0e38f; l_i[i] = 0.0f; }
  const float scale = 0.0625f;   // 1/sqrt(2*128)

  for (int jb = 0; jb < L_ / 64; ++jb) {
    const int jBase = jb * 64;
    __syncthreads();             // previous tile's LDS reads complete
    // Async DMA: K / KF tiles straight into LDS (no VGPR bounce)
    for (int i = 0; i < 4; ++i) {
      int c = tid + i * 256;
      int j = c >> 4, dc = (c & 15) << 3;
      g2l_async_b128(lds_off(&lK [j * 128 + dc]), &k [(jBase + j) * 128 + dc]);
      g2l_async_b128(lds_off(&lKF[j * 128 + dc]), &kf[(jBase + j) * 128 + dc]);
    }
    // V transposed into LDS via VGPRs; overlaps with the in-flight asyncs
    for (int i = 0; i < 4; ++i) {
      int c = tid + i * 256;
      int j = c >> 4, dc = (c & 15) << 3;
      uint4 vv4 = *(const uint4*)&v[(jBase + j) * 128 + dc];
      const unsigned short* pv = (const unsigned short*)&vv4;
      for (int e = 0; e < 8; ++e) lVt[(dc + e) * 64 + j] = pv[e];
    }
    wait_async<0>();
    __syncthreads();

    // S = (Q K^T + QF KF^T) * scale    (16 rows x 64 key cols per wave)
    v8f s[4];
    for (int jt = 0; jt < 4; ++jt) {
      v8f a = {};
      int jrow = jt * 16 + l16;
      for (int kd = 0; kd < 4; ++kd) {
        FragB16 bK;
        bK.u[0] = *(const uint4*)&lK[jrow * 128 + kd * 32 + half * 8];
        bK.u[1] = *(const uint4*)&lK[jrow * 128 + kd * 32 + 16 + half * 8];
        a = WMMA_BF16(qA[kd].v, bK.v, a);
      }
      for (int kd = 0; kd < 4; ++kd) {
        FragB16 bF;
        bF.u[0] = *(const uint4*)&lKF[jrow * 128 + kd * 32 + half * 8];
        bF.u[1] = *(const uint4*)&lKF[jrow * 128 + kd * 32 + 16 + half * 8];
        a = WMMA_BF16(qfA[kd].v, bF.v, a);
      }
      for (int e = 0; e < 8; ++e) a[e] *= scale;
      s[jt] = a;
    }

    // Online softmax. Row r = vv + 8*half lives across lanes of its 16-lane
    // half, so xor-shuffles with masks 1,2,4,8 reduce exactly one row.
    for (int vv = 0; vv < 8; ++vv) {
      float mx = fmaxf(fmaxf(s[0][vv], s[1][vv]), fmaxf(s[2][vv], s[3][vv]));
      for (int off = 1; off < 16; off <<= 1) mx = fmaxf(mx, __shfl_xor(mx, off, 32));
      float mnew  = fmaxf(m_i[vv], mx);
      float alpha = __expf(m_i[vv] - mnew);
      float rs = 0.0f;
      for (int jt = 0; jt < 4; ++jt) {
        float p = __expf(s[jt][vv] - mnew);
        s[jt][vv] = p;
        rs += p;
      }
      for (int off = 1; off < 16; off <<= 1) rs += __shfl_xor(rs, off, 32);
      l_i[vv] = l_i[vv] * alpha + rs;
      m_i[vv] = mnew;
      for (int dT = 0; dT < 8; ++dT) oacc[dT][vv] *= alpha;
    }

    // Re-layout P (C-layout) -> A-fragment via per-wave LDS scratch
    const int wbase = wave * (16 * 64);
    for (int jt = 0; jt < 4; ++jt)
      for (int vv = 0; vv < 8; ++vv)
        lP[wbase + (vv + half * 8) * 64 + jt * 16 + l16] = f2bf(s[jt][vv]);
    asm volatile("s_wait_dscnt 0" ::: "memory");   // wave-local store->load

    FragB16 pA[2];
    for (int kp = 0; kp < 2; ++kp) {
      pA[kp].u[0] = *(const uint4*)&lP[wbase + l16 * 64 + kp * 32 + half * 8];
      pA[kp].u[1] = *(const uint4*)&lP[wbase + l16 * 64 + kp * 32 + 16 + half * 8];
    }
    for (int dT = 0; dT < 8; ++dT)
      for (int kp = 0; kp < 2; ++kp) {
        FragB16 bV;
        int d = dT * 16 + l16;
        bV.u[0] = *(const uint4*)&lVt[d * 64 + kp * 32 + half * 8];
        bV.u[1] = *(const uint4*)&lVt[d * 64 + kp * 32 + 16 + half * 8];
        oacc[dT] = WMMA_BF16(pA[kp].v, bV.v, oacc[dT]);
      }
  }

  // Normalize and store (flat identity reshape back to (B,L,DF))
  for (int dT = 0; dT < 8; ++dT) {
    int d = dT * 16 + l16;
    for (int vv = 0; vv < 8; ++vv) {
      int row = mBase + wave * 16 + vv + half * 8;
      o[row * 128 + d] = f2bf(oacc[dT][vv] / l_i[vv]);
    }
  }
}

// ---------------------------------------------------------------------------
extern "C" void kernel_launch(void* const* d_in, const int* in_sizes, int n_in,
                              void* d_out, int out_size, void* d_ws, size_t ws_size,
                              hipStream_t stream) {
  (void)in_sizes; (void)n_in; (void)out_size; (void)ws_size;
  const float* x    = (const float*)d_in[0];
  const float* xk   = (const float*)d_in[1];
  const float* Wv   = (const float*)d_in[2];  const float* bv   = (const float*)d_in[3];
  const float* Wk   = (const float*)d_in[4];  const float* bk   = (const float*)d_in[5];
  const float* Wq   = (const float*)d_in[6];  const float* bq   = (const float*)d_in[7];
  const float* Wkk  = (const float*)d_in[8];  const float* bkk  = (const float*)d_in[9];
  const float* Wkq  = (const float*)d_in[10]; const float* bkq  = (const float*)d_in[11];
  const float* Wq2  = (const float*)d_in[12]; const float* bq2  = (const float*)d_in[13];
  const float* Wk2  = (const float*)d_in[14]; const float* bk2  = (const float*)d_in[15];
  const float* Wqf  = (const float*)d_in[16]; const float* bqf  = (const float*)d_in[17];
  const float* Wkf  = (const float*)d_in[18]; const float* bkf  = (const float*)d_in[19];
  const float* Wout = (const float*)d_in[20]; const float* bout = (const float*)d_in[21];
  float* out = (float*)d_out;

  char* ws = (char*)d_ws;
  size_t off = 0;
  auto alloc = [&](size_t elems) -> unsigned short* {
    unsigned short* p = (unsigned short*)(ws + off);
    off = (off + elems * sizeof(unsigned short) + 255) & ~(size_t)255;
    return p;
  };

  unsigned short* xb   = alloc((size_t)M_ * DM_);
  unsigned short* xkb  = alloc((size_t)M_ * DM_);
  unsigned short* wvB  = alloc((size_t)DF_ * DM_);
  unsigned short* wkB  = alloc((size_t)DF_ * DM_);
  unsigned short* wqB  = alloc((size_t)DF_ * DM_);
  unsigned short* wkkB = alloc((size_t)DF_ * DM_);
  unsigned short* wkqB = alloc((size_t)DF_ * DM_);
  unsigned short* wq2B = alloc((size_t)DF_ * DF_);
  unsigned short* wk2B = alloc((size_t)DF_ * DF_);
  unsigned short* wqfB = alloc((size_t)DF_ * DF_);
  unsigned short* wkfB = alloc((size_t)DF_ * DF_);
  unsigned short* woB  = alloc((size_t)DM_ * DF_);
  unsigned short* v1   = alloc((size_t)M_ * DF_);
  unsigned short* k1   = alloc((size_t)M_ * DF_);
  unsigned short* q1   = alloc((size_t)M_ * DF_);
  unsigned short* kf1  = alloc((size_t)M_ * DF_);
  unsigned short* qf1  = alloc((size_t)M_ * DF_);
  unsigned short* q2   = alloc((size_t)M_ * DF_);
  unsigned short* k2   = alloc((size_t)M_ * DF_);
  unsigned short* qf2  = alloc((size_t)M_ * DF_);
  unsigned short* kf2  = alloc((size_t)M_ * DF_);
  unsigned short* attO = alloc((size_t)M_ * DF_);

  // 1) f32 -> bf16 conversions
  struct CvtJob { const float* s; unsigned short* d; int n; };
  const CvtJob jobs[] = {
    {x, xb, M_ * DM_}, {xk, xkb, M_ * DM_},
    {Wv, wvB, DF_ * DM_}, {Wk, wkB, DF_ * DM_}, {Wq, wqB, DF_ * DM_},
    {Wkk, wkkB, DF_ * DM_}, {Wkq, wkqB, DF_ * DM_},
    {Wq2, wq2B, DF_ * DF_}, {Wk2, wk2B, DF_ * DF_},
    {Wqf, wqfB, DF_ * DF_}, {Wkf, wkfB, DF_ * DF_},
    {Wout, woB, DM_ * DF_},
  };
  for (const CvtJob& j : jobs) {
    int n4 = j.n / 4;
    cvt_f32_bf16<<<(n4 + 255) / 256, 256, 0, stream>>>(j.s, j.d, n4);
  }

  // 2) Stage-1 projections (M=4096, N=1024, K=512)
  dim3 g1(DF_ / 128, M_ / 64), blk(256);
  gemm_bf16<<<g1, blk, 0, stream>>>(xb,  wvB,  bv,  v1,  nullptr, M_, DF_, DM_);
  gemm_bf16<<<g1, blk, 0, stream>>>(xb,  wkB,  bk,  k1,  nullptr, M_, DF_, DM_);
  gemm_bf16<<<g1, blk, 0, stream>>>(xb,  wqB,  bq,  q1,  nullptr, M_, DF_, DM_);
  gemm_bf16<<<g1, blk, 0, stream>>>(xkb, wkkB, bkk, kf1, nullptr, M_, DF_, DM_);
  gemm_bf16<<<g1, blk, 0, stream>>>(xkb, wkqB, bkq, qf1, nullptr, M_, DF_, DM_);

  // 3) Stage-2 projections (M=4096, N=1024, K=1024)
  gemm_bf16<<<g1, blk, 0, stream>>>(q1,  wq2B, bq2, q2,  nullptr, M_, DF_, DF_);
  gemm_bf16<<<g1, blk, 0, stream>>>(k1,  wk2B, bk2, k2,  nullptr, M_, DF_, DF_);
  gemm_bf16<<<g1, blk, 0, stream>>>(qf1, wqfB, bqf, qf2, nullptr, M_, DF_, DF_);
  gemm_bf16<<<g1, blk, 0, stream>>>(kf1, wkfB, bkf, kf2, nullptr, M_, DF_, DF_);

  // 4) Fused dual-score attention over the 32 contiguous (b,h) slabs
  dim3 ga(L_ / 128, B_ * H_);
  attn_fused<<<ga, blk, 0, stream>>>(q2, k2, qf2, kf2, v1, attO);

  // 5) Output projection (M=4096, N=512, K=1024) -> f32 d_out
  dim3 g2(DM_ / 128, M_ / 64);
  gemm_bf16<<<g2, blk, 0, stream>>>(attO, woB, bout, nullptr, out, M_, DM_, DF_);
}
